// HybridGNNAnomalyDetector_19902878450310
// MI455X (gfx1250) — matrix operand order repaired and checked by reading, not compiled, not run on hardware
//
#include <hip/hip_runtime.h>
#include <hip/hip_bf16.h>
#include <math.h>

// ---------------- problem constants ----------------
#define Nn      30000
#define Ee      480000
#define E2C     510000      // Ee + Nn self loops
#define DN      256
#define DE      32
#define HIDC    64
#define EMBD    32
#define NHEADS  4
#define NEG_SLOPE 0.2f
#define BN_INV  0.9999950000374997f   // 1/sqrt(1+1e-5)

typedef __attribute__((ext_vector_type(16))) _Float16 v16h;
typedef __attribute__((ext_vector_type(8)))  float    v8f;

// ---------------- WMMA helpers (gfx1250, wave32) ----------------
__device__ inline v8f wmma_f16(v16h a, v16h b, v8f c) {
  return __builtin_amdgcn_wmma_f32_16x16x32_f16(false, a, false, b, (short)0, c, false, false);
}

// A 16x32 f16 fragment from a row pointer already offset by (k0 + kh):
// elements 0..7  = p[0..7], elements 8..15 = p[16..23]  (per ISA A layout)
__device__ inline v16h packA(const float* p) {
  const float4 x0 = *(const float4*)(p);
  const float4 x1 = *(const float4*)(p + 4);
  const float4 y0 = *(const float4*)(p + 16);
  const float4 y1 = *(const float4*)(p + 20);
  v16h a;
  a[0]=(_Float16)x0.x; a[1]=(_Float16)x0.y; a[2]=(_Float16)x0.z; a[3]=(_Float16)x0.w;
  a[4]=(_Float16)x1.x; a[5]=(_Float16)x1.y; a[6]=(_Float16)x1.z; a[7]=(_Float16)x1.w;
  a[8]=(_Float16)y0.x; a[9]=(_Float16)y0.y; a[10]=(_Float16)y0.z; a[11]=(_Float16)y0.w;
  a[12]=(_Float16)y1.x; a[13]=(_Float16)y1.y; a[14]=(_Float16)y1.z; a[15]=(_Float16)y1.w;
  return a;
}

// B 32x16 f16 fragment: lane l holds col n0+l%16; elements 0..7 = K-rows k0+kh+0..7, 8..15 = k0+kh+16..23 (kh = 8*(l/16))
__device__ inline v16h load_Bh(const _Float16* W, int ldn, int k0, int n0) {
  int lane = threadIdx.x & 31;
  const _Float16* p = W + (size_t)(k0 + (lane >> 4) * 8) * ldn + n0 + (lane & 15);
  v16h b;
#pragma unroll
  for (int s = 0; s < 8; ++s) { b[s] = p[(size_t)s * ldn]; b[s + 8] = p[(size_t)(s + 16) * ldn]; }
  return b;
}

// ---------------- f32 -> f16 weight conversion ----------------
__global__ void f2h_kernel(const float* s, _Float16* d, int n) {
  int i = blockIdx.x * blockDim.x + threadIdx.x;
  if (i < n) d[i] = (_Float16)s[i];
}

// ---------------- register-tiled GEMM: Y = X @ W (NT 16-col tiles / wave) ----------------
template <int NT>
__global__ __launch_bounds__(128) void gemm_wmma_kernel(
    const float* X, const _Float16* Wh, float* Y, int M, int K, int Nout) {
  int wave = threadIdx.x >> 5, lane = threadIdx.x & 31;
  int row0 = (blockIdx.x * 4 + wave) * 16;
  int n0   = blockIdx.y * (16 * NT);
  if (row0 >= M) return;
  int r  = lane & 15;
  int kh = (lane >> 4) * 8;
  int gr = row0 + r; if (gr >= M) gr = M - 1;      // clamp: keeps loads vectorized
  const float* rowp = X + (size_t)gr * K + kh;
  v8f acc[NT];
#pragma unroll
  for (int t = 0; t < NT; ++t) acc[t] = (v8f){};
  for (int k0 = 0; k0 < K; k0 += 32) {
    v16h a = packA(rowp + k0);
#pragma unroll
    for (int t = 0; t < NT; ++t)
      acc[t] = wmma_f16(a, load_Bh(Wh, Nout, k0, n0 + t * 16), acc[t]);
  }
  int col = lane & 15, roff = (lane >> 4) * 8;
  if (row0 + 16 <= M) {                            // fast path: whole tile in range
#pragma unroll
    for (int t = 0; t < NT; ++t)
#pragma unroll
      for (int rr = 0; rr < 8; ++rr)
        Y[(size_t)(row0 + rr + roff) * Nout + n0 + t * 16 + col] = acc[t][rr];
  } else {
#pragma unroll
    for (int t = 0; t < NT; ++t)
#pragma unroll
      for (int rr = 0; rr < 8; ++rr) {
        int gw = row0 + rr + roff;
        if (gw < M) Y[(size_t)gw * Nout + n0 + t * 16 + col] = acc[t][rr];
      }
  }
}

// ---------------- fused 3-layer MLP (Lin-ReLU-BN-Lin-ReLU-Lin), WMMA + LDS ----------------
template <int KIN, int H1, int H2, int NOUT, bool SIG, bool EDGE>
__global__ __launch_bounds__(128) void mlp_kernel(
    const float* Xin, const int* srcA, const int* dstA, const float* eattr, int M,
    const _Float16* W1h, const float* b1v, const float* g1v, const float* be1v,
    const _Float16* W2h, const float* b2v,
    const float* W3f, const _Float16* W3h, const float* b3v, float* out) {
  __shared__ float sh1[4 * 16 * H1];
  __shared__ float sh2[4 * 16 * H2];
  int wave = threadIdx.x >> 5, lane = threadIdx.x & 31;
  int row0 = (blockIdx.x * 4 + wave) * 16;
  float* s1 = &sh1[wave * 16 * H1];
  float* s2 = &sh2[wave * 16 * H2];
  int col = lane & 15, roff = (lane >> 4) * 8;
  int r = lane & 15, kh = (lane >> 4) * 8;

  // ---- hoisted input fragments (loaded once, reused across all n-chunks) ----
  constexpr int NKA = KIN / 32;
  v16h a1[NKA];
  if constexpr (EDGE) {
    int e = row0 + r; if (e >= M) e = M - 1;       // clamp
    const float* b0 = Xin   + (size_t)srcA[e] * EMBD + kh;   // z[src]
    const float* b1 = Xin   + (size_t)dstA[e] * EMBD + kh;   // z[dst]
    const float* b2 = eattr + (size_t)e       * DE   + kh;   // edge_attr
    a1[0] = packA(b0); a1[1] = packA(b1); a1[2] = packA(b2);
  } else {
    int gr = row0 + r; if (gr >= M) gr = M - 1;
    const float* p = Xin + (size_t)gr * KIN + kh;
#pragma unroll
    for (int ks = 0; ks < NKA; ++ks) a1[ks] = packA(p + ks * 32);
  }

  // layer 1: KIN -> H1, ReLU + eval-BN
  for (int nc = 0; nc < H1 / 16; ++nc) {
    v8f acc = {};
#pragma unroll
    for (int ks = 0; ks < NKA; ++ks)
      acc = wmma_f16(a1[ks], load_Bh(W1h, H1, ks * 32, nc * 16), acc);
    int n = nc * 16 + col;
    float bb = b1v[n], gg = g1v[n], be = be1v[n];
#pragma unroll
    for (int rr = 0; rr < 8; ++rr) {
      float v = acc[rr] + bb;  v = v > 0.f ? v : 0.f;
      s1[(rr + roff) * H1 + n] = gg * (v * BN_INV) + be;
    }
  }
  __syncthreads();

  // layer 2: H1 -> H2, ReLU
  v16h a2[H1 / 32];
  {
    const float* p = s1 + r * H1 + kh;
#pragma unroll
    for (int ks = 0; ks < H1 / 32; ++ks) a2[ks] = packA(p + ks * 32);
  }
  for (int nc = 0; nc < H2 / 16; ++nc) {
    v8f acc = {};
#pragma unroll
    for (int ks = 0; ks < H1 / 32; ++ks)
      acc = wmma_f16(a2[ks], load_Bh(W2h, H2, ks * 32, nc * 16), acc);
    int n = nc * 16 + col;
    float bb = b2v[n];
#pragma unroll
    for (int rr = 0; rr < 8; ++rr) {
      float v = acc[rr] + bb;
      s2[(rr + roff) * H2 + n] = v > 0.f ? v : 0.f;
    }
  }
  __syncthreads();

  // layer 3: H2 -> NOUT
  if constexpr (NOUT >= 16) {
    v16h a3[H2 / 32];
    const float* p = s2 + r * H2 + kh;
#pragma unroll
    for (int ks = 0; ks < H2 / 32; ++ks) a3[ks] = packA(p + ks * 32);
    bool full = (row0 + 16 <= M);
    for (int nc = 0; nc < NOUT / 16; ++nc) {
      v8f acc = {};
#pragma unroll
      for (int ks = 0; ks < H2 / 32; ++ks)
        acc = wmma_f16(a3[ks], load_Bh(W3h, NOUT, ks * 32, nc * 16), acc);
      int n = nc * 16 + col;
      float bb = b3v[n];
      if (full) {                                  // fast path: unpredicated stores
#pragma unroll
        for (int rr = 0; rr < 8; ++rr)
          out[(size_t)(row0 + rr + roff) * NOUT + n] = acc[rr] + bb;
      } else {
#pragma unroll
        for (int rr = 0; rr < 8; ++rr) {
          int gw = row0 + rr + roff;
          if (gw < M) out[(size_t)gw * NOUT + n] = acc[rr] + bb;
        }
      }
    }
  } else {   // scalar output head: VALU dot
    if (lane < 16) {
      float s = b3v[0];
      for (int j = 0; j < H2; ++j) s += s2[lane * H2 + j] * W3f[j];
      if constexpr (SIG) s = 1.f / (1.f + expf(-s));
      int gw = row0 + lane;
      if (gw < M) out[gw] = s;
    }
  }
}

// ---------------- sortable-uint float max for atomics ----------------
__device__ inline unsigned f2sort(float f) {
  unsigned u = __float_as_uint(f);
  return (u & 0x80000000u) ? ~u : (u | 0x80000000u);
}
__device__ inline float sort2f(unsigned u) {
  return (u & 0x80000000u) ? __uint_as_float(u & 0x7FFFFFFFu) : __uint_as_float(~u);
}

// ---------------- small utility kernels ----------------
__global__ void zero_kernel(float* p, size_t n) {
  for (size_t i = blockIdx.x * (size_t)blockDim.x + threadIdx.x; i < n;
       i += (size_t)gridDim.x * blockDim.x) p[i] = 0.f;
}
__global__ void colmean_kernel(const float* X, float* out, int rows, int ld) {
  __shared__ float red[256];
  int c = blockIdx.x; float s = 0.f;
  for (int r = threadIdx.x; r < rows; r += blockDim.x) s += X[(size_t)r * ld + c];
  red[threadIdx.x] = s; __syncthreads();
  for (int o = 128; o > 0; o >>= 1) {
    if ((int)threadIdx.x < o) red[threadIdx.x] += red[threadIdx.x + o];
    __syncthreads();
  }
  if (threadIdx.x == 0) out[c] = red[0] / (float)rows;
}
// ve[h][d] = sum_c We[d, h*C+c] * att_edge[h*C+c]
__global__ void ve_kernel(const float* We, const float* att_e, float* ve, int H, int C) {
  int t = blockIdx.x * blockDim.x + threadIdx.x;
  if (t >= H * DE) return;
  int h = t / DE, d = t - h * DE;
  float s = 0.f;
  for (int c = 0; c < C; ++c) s += We[(size_t)d * H * C + h * C + c] * att_e[h * C + c];
  ve[h * DE + d] = s;
}
__global__ void node_att_kernel(const float* xh, const float* att_s, const float* att_d,
                                float* asrc, float* adst, int H, int C) {
  int t = blockIdx.x * blockDim.x + threadIdx.x;
  if (t >= Nn * H) return;
  int n = t / H, h = t - n * H;
  const float* row = xh + ((size_t)n * H + h) * C;
  float s = 0.f, dd = 0.f;
  for (int c = 0; c < C; ++c) { float v = row[c]; s += v * att_s[h * C + c]; dd += v * att_d[h * C + c]; }
  asrc[t] = s; adst[t] = dd;
}
__global__ void ae_kernel(const float* eattr, const float* eamean, const float* ve,
                          float* ae, int H) {
  int t = blockIdx.x * blockDim.x + threadIdx.x;
  if (t >= E2C * H) return;
  int e2 = t / H, h = t - e2 * H;
  const float* row = (e2 < Ee) ? eattr + (size_t)e2 * DE : eamean;
  float s = 0.f;
  for (int d = 0; d < DE; ++d) s += row[d] * ve[h * DE + d];
  ae[t] = s;
}
__global__ void alpha_kernel(const int* srcA, const int* dstA, const float* asrc,
                             const float* adst, const float* ae, float* alpha,
                             unsigned* amax, int H) {
  int t = blockIdx.x * blockDim.x + threadIdx.x;
  if (t >= E2C * H) return;
  int e2 = t / H, h = t - e2 * H;
  int s, d;
  if (e2 < Ee) { s = srcA[e2]; d = dstA[e2]; } else { s = d = e2 - Ee; }
  float a = asrc[s * H + h] + adst[d * H + h] + ae[t];
  a = a >= 0.f ? a : NEG_SLOPE * a;        // leaky relu
  alpha[t] = a;
  atomicMax(&amax[d * H + h], f2sort(a));  // segment max
}
__global__ void exp_kernel(const int* dstA, float* alpha, const unsigned* amax,
                           float* den, int H) {
  int t = blockIdx.x * blockDim.x + threadIdx.x;
  if (t >= E2C * H) return;
  int e2 = t / H, h = t - e2 * H;
  int d = (e2 < Ee) ? dstA[e2] : (e2 - Ee);
  float ex = expf(alpha[t] - sort2f(amax[d * H + h]));
  alpha[t] = ex;                            // reuse buffer for ex
  atomicAdd(&den[d * H + h], ex);
}
__global__ void agg_kernel(const int* srcA, const int* dstA, const float* xh,
                           const float* ex, const float* den, float* outb, int H, int C) {
  int t = blockIdx.x * blockDim.x + threadIdx.x;
  if (t >= E2C * H) return;
  int e2 = t / H, h = t - e2 * H;
  int s, d;
  if (e2 < Ee) { s = srcA[e2]; d = dstA[e2]; } else { s = d = e2 - Ee; }
  float coef = ex[t] / (den[d * H + h] + 1e-16f);
  const float* ms = xh + ((size_t)s * H + h) * C;
  float* od = outb + ((size_t)d * H + h) * C;
  __builtin_prefetch(ms, 0, 0);             // global_prefetch_b8
  for (int c = 0; c < C; ++c) atomicAdd(&od[c], ms[c] * coef);
}
// out0 -> h : +bias, BN, ELU, +x (in place)
__global__ void post0_kernel(float* out0, const float* x, const float* bias,
                             const float* gamma, const float* beta, size_t n) {
  for (size_t i = blockIdx.x * (size_t)blockDim.x + threadIdx.x; i < n;
       i += (size_t)gridDim.x * blockDim.x) {
    int col = (int)(i % DN);
    float v = out0[i] + bias[col];
    v = gamma[col] * (v * BN_INV) + beta[col];
    v = v > 0.f ? v : expm1f(v);
    out0[i] = v + x[i];
  }
}
__global__ void zbias_kernel(float* z, const float* bias, size_t n) {
  for (size_t i = blockIdx.x * (size_t)blockDim.x + threadIdx.x; i < n;
       i += (size_t)gridDim.x * blockDim.x) z[i] += bias[i & (EMBD - 1)];
}
// tiny 1-row MLP for global head: 32 -> 64 -> 32 -> 1
__global__ void glob_kernel(const float* g, const float* W1, const float* b1,
                            const float* g1, const float* be1, const float* W2,
                            const float* b2, const float* W3, const float* b3, float* out) {
  __shared__ float s1[64]; __shared__ float s2[32];
  int t = threadIdx.x;
  if (t < 64) {
    float a = b1[t];
    for (int j = 0; j < 32; ++j) a += g[j] * W1[j * 64 + t];
    a = a > 0.f ? a : 0.f;
    s1[t] = g1[t] * (a * BN_INV) + be1[t];
  }
  __syncthreads();
  if (t < 32) {
    float a = b2[t];
    for (int j = 0; j < 64; ++j) a += s1[j] * W2[j * 32 + t];
    s2[t] = a > 0.f ? a : 0.f;
  }
  __syncthreads();
  if (t == 0) {
    float a = b3[0];
    for (int j = 0; j < 32; ++j) a += s2[j] * W3[j];
    out[0] = a;
  }
}

// ---------------- workspace layout (floats) ----------------
static const size_t F_XH0  = 0;                       // 30000*256
static const size_t F_OUT0 = F_XH0  + 7680000;        // agg0, becomes h in place
static const size_t F_AS0  = F_OUT0 + 7680000;        // 30000*4
static const size_t F_AD0  = F_AS0  + 120000;
static const size_t F_AE0  = F_AD0  + 120000;         // 510000*4
static const size_t F_AL0  = F_AE0  + 2040000;        // alpha0 / ex0
static const size_t F_AMX0 = F_AL0  + 2040000;        // uint
static const size_t F_DEN0 = F_AMX0 + 120000;
static const size_t F_XH1  = F_DEN0 + 120000;         // 30000*32
static const size_t F_AS1  = F_XH1  + 960000;
static const size_t F_AD1  = F_AS1  + 30000;
static const size_t F_AE1  = F_AD1  + 30000;          // 510000
static const size_t F_AL1  = F_AE1  + 510000;
static const size_t F_AMX1 = F_AL1  + 510000;
static const size_t F_DEN1 = F_AMX1 + 30000;
static const size_t F_VE0  = F_DEN1 + 30000;          // 4*32
static const size_t F_VE1  = F_VE0  + 128;            // 32
static const size_t F_EAM  = F_VE1  + 32;             // 32
static const size_t F_H16  = F_EAM  + 32;             // f16 weight pool starts here

// f16 weight pool offsets (halves, all multiples of 16 -> 32B aligned)
static const size_t H_G0W  = 0;        // 256*256
static const size_t H_G1W  = 65536;    // 256*32
static const size_t H_NDW1 = 73728;    // 32*128
static const size_t H_NDW2 = 77824;    // 128*64
static const size_t H_NDW3 = 86016;    // 64*256
static const size_t H_NAW1 = 102400;   // 32*64
static const size_t H_NAW2 = 104448;   // 64*32
static const size_t H_EDW1 = 106496;   // 96*128
static const size_t H_EDW2 = 118784;   // 128*64
static const size_t H_EAW1 = 126976;   // 96*64
static const size_t H_EAW2 = 133120;   // 64*32

// output offsets (reference return order)
static const size_t O_NS = 0;              // node_scores [30000]
static const size_t O_ES = 30000;          // edge_scores [480000]
static const size_t O_GS = 510000;         // global_score [1]
static const size_t O_NR = 510001;         // node_recon [30000*256]
static const size_t O_ER = 8190001;        // edge_recon [480000]
static const size_t O_Z  = 8670001;        // z [30000*32]
static const size_t O_G  = 9630001;        // g [32]

extern "C" void kernel_launch(void* const* d_in, const int* in_sizes, int n_in,
                              void* d_out, int out_size, void* d_ws, size_t ws_size,
                              hipStream_t stream) {
  (void)in_sizes; (void)n_in; (void)out_size; (void)ws_size;
  // ---- inputs (JAX pytree flattening: dict keys sorted) ----
  const float* x    = (const float*)d_in[0];
  const int*   ei   = (const int*)  d_in[1];
  const float* eat  = (const float*)d_in[2];
  const float* bn0_beta  = (const float*)d_in[3];
  const float* bn0_gamma = (const float*)d_in[4];
  const float *eaW1=(const float*)d_in[5],  *eaW2=(const float*)d_in[6],  *eaW3=(const float*)d_in[7];
  const float *eab1=(const float*)d_in[8],  *eab2=(const float*)d_in[9],  *eab3=(const float*)d_in[10];
  const float *eabe1=(const float*)d_in[11],*eag1=(const float*)d_in[12];
  const float *edW1=(const float*)d_in[13], *edW2=(const float*)d_in[14], *edW3=(const float*)d_in[15];
  const float *edb1=(const float*)d_in[16], *edb2=(const float*)d_in[17], *edb3=(const float*)d_in[18];
  const float *edbe1=(const float*)d_in[19],*edg1=(const float*)d_in[20];
  const float *g0W=(const float*)d_in[21],  *g0We=(const float*)d_in[22];
  const float *g0ad=(const float*)d_in[23], *g0ae=(const float*)d_in[24], *g0as=(const float*)d_in[25];
  const float *g0b=(const float*)d_in[26];
  const float *g1W=(const float*)d_in[27],  *g1We=(const float*)d_in[28];
  const float *g1ad=(const float*)d_in[29], *g1ae=(const float*)d_in[30], *g1as=(const float*)d_in[31];
  const float *g1b=(const float*)d_in[32];
  const float *gaW1=(const float*)d_in[33], *gaW2=(const float*)d_in[34], *gaW3=(const float*)d_in[35];
  const float *gab1=(const float*)d_in[36], *gab2=(const float*)d_in[37], *gab3=(const float*)d_in[38];
  const float *gabe1=(const float*)d_in[39],*gag1=(const float*)d_in[40];
  const float *naW1=(const float*)d_in[41], *naW2=(const float*)d_in[42], *naW3=(const float*)d_in[43];
  const float *nab1=(const float*)d_in[44], *nab2=(const float*)d_in[45], *nab3=(const float*)d_in[46];
  const float *nabe1=(const float*)d_in[47],*nag1=(const float*)d_in[48];
  const float *ndW1=(const float*)d_in[49], *ndW2=(const float*)d_in[50], *ndW3=(const float*)d_in[51];
  const float *ndb1=(const float*)d_in[52], *ndb2=(const float*)d_in[53], *ndb3=(const float*)d_in[54];
  const float *ndbe1=(const float*)d_in[55],*ndg1=(const float*)d_in[56];

  const int* srcA = ei;
  const int* dstA = ei + Ee;

  float* ws = (float*)d_ws;
  float* outp = (float*)d_out;
  float *XH0 = ws + F_XH0, *OUT0 = ws + F_OUT0;
  float *AS0 = ws + F_AS0, *AD0 = ws + F_AD0, *AE0 = ws + F_AE0, *AL0 = ws + F_AL0;
  unsigned* AMX0 = (unsigned*)(ws + F_AMX0);
  float *DEN0 = ws + F_DEN0, *XH1 = ws + F_XH1;
  float *AS1 = ws + F_AS1, *AD1 = ws + F_AD1, *AE1 = ws + F_AE1, *AL1 = ws + F_AL1;
  unsigned* AMX1 = (unsigned*)(ws + F_AMX1);
  float *DEN1 = ws + F_DEN1, *VE0 = ws + F_VE0, *VE1 = ws + F_VE1, *EAM = ws + F_EAM;
  _Float16* hw = (_Float16*)(ws + F_H16);
  _Float16 *HG0W=hw+H_G0W, *HG1W=hw+H_G1W, *HND1=hw+H_NDW1, *HND2=hw+H_NDW2, *HND3=hw+H_NDW3;
  _Float16 *HNA1=hw+H_NAW1, *HNA2=hw+H_NAW2, *HED1=hw+H_EDW1, *HED2=hw+H_EDW2;
  _Float16 *HEA1=hw+H_EAW1, *HEA2=hw+H_EAW2;
  float *NS = outp + O_NS, *ES = outp + O_ES, *GS = outp + O_GS;
  float *NR = outp + O_NR, *ER = outp + O_ER, *Z = outp + O_Z, *G = outp + O_G;

  // ---- init accumulators (every launch; deterministic) ----
  zero_kernel<<<4096, 256, 0, stream>>>(OUT0, (size_t)7680000);
  zero_kernel<<<256,  256, 0, stream>>>((float*)AMX0, (size_t)120000);
  zero_kernel<<<256,  256, 0, stream>>>(DEN0, (size_t)120000);
  zero_kernel<<<64,   256, 0, stream>>>((float*)AMX1, (size_t)30000);
  zero_kernel<<<64,   256, 0, stream>>>(DEN1, (size_t)30000);
  zero_kernel<<<1024, 256, 0, stream>>>(Z, (size_t)960000);

  // ---- convert WMMA weights to f16 ----
  f2h_kernel<<<256, 256, 0, stream>>>(g0W,  HG0W, 65536);
  f2h_kernel<<<32,  256, 0, stream>>>(g1W,  HG1W, 8192);
  f2h_kernel<<<16,  256, 0, stream>>>(ndW1, HND1, 4096);
  f2h_kernel<<<32,  256, 0, stream>>>(ndW2, HND2, 8192);
  f2h_kernel<<<64,  256, 0, stream>>>(ndW3, HND3, 16384);
  f2h_kernel<<<8,   256, 0, stream>>>(naW1, HNA1, 2048);
  f2h_kernel<<<8,   256, 0, stream>>>(naW2, HNA2, 2048);
  f2h_kernel<<<48,  256, 0, stream>>>(edW1, HED1, 12288);
  f2h_kernel<<<32,  256, 0, stream>>>(edW2, HED2, 8192);
  f2h_kernel<<<24,  256, 0, stream>>>(eaW1, HEA1, 6144);
  f2h_kernel<<<8,   256, 0, stream>>>(eaW2, HEA2, 2048);

  // ---- edge-attr mean (self-loop fill), edge-attention projections ----
  colmean_kernel<<<DE, 256, 0, stream>>>(eat, EAM, Ee, DE);
  ve_kernel<<<1, 128, 0, stream>>>(g0We, g0ae, VE0, NHEADS, HIDC);
  ve_kernel<<<1, 32,  0, stream>>>(g1We, g1ae, VE1, 1, EMBD);

  // ---- GAT layer 0 ----
  gemm_wmma_kernel<4><<<dim3(469, 4), 128, 0, stream>>>(x, HG0W, XH0, Nn, DN, DN);
  node_att_kernel<<<(Nn * NHEADS + 255) / 256, 256, 0, stream>>>(XH0, g0as, g0ad, AS0, AD0, NHEADS, HIDC);
  int t0 = E2C * NHEADS;
  ae_kernel   <<<(t0 + 255) / 256, 256, 0, stream>>>(eat, EAM, VE0, AE0, NHEADS);
  alpha_kernel<<<(t0 + 255) / 256, 256, 0, stream>>>(srcA, dstA, AS0, AD0, AE0, AL0, AMX0, NHEADS);
  exp_kernel  <<<(t0 + 255) / 256, 256, 0, stream>>>(dstA, AL0, AMX0, DEN0, NHEADS);
  agg_kernel  <<<(t0 + 255) / 256, 256, 0, stream>>>(srcA, dstA, XH0, AL0, DEN0, OUT0, NHEADS, HIDC);
  post0_kernel<<<4096, 256, 0, stream>>>(OUT0, x, g0b, bn0_gamma, bn0_beta, (size_t)7680000);

  // ---- GAT layer 1 ----
  gemm_wmma_kernel<2><<<dim3(469, 1), 128, 0, stream>>>(OUT0, HG1W, XH1, Nn, DN, EMBD);
  node_att_kernel<<<(Nn + 255) / 256, 256, 0, stream>>>(XH1, g1as, g1ad, AS1, AD1, 1, EMBD);
  ae_kernel   <<<(E2C + 255) / 256, 256, 0, stream>>>(eat, EAM, VE1, AE1, 1);
  alpha_kernel<<<(E2C + 255) / 256, 256, 0, stream>>>(srcA, dstA, AS1, AD1, AE1, AL1, AMX1, 1);
  exp_kernel  <<<(E2C + 255) / 256, 256, 0, stream>>>(dstA, AL1, AMX1, DEN1, 1);
  agg_kernel  <<<(E2C + 255) / 256, 256, 0, stream>>>(srcA, dstA, XH1, AL1, DEN1, Z, 1, EMBD);
  zbias_kernel<<<1024, 256, 0, stream>>>(Z, g1b, (size_t)960000);

  // ---- global pooling + global head ----
  colmean_kernel<<<EMBD, 256, 0, stream>>>(Z, G, Nn, EMBD);
  glob_kernel<<<1, 64, 0, stream>>>(G, gaW1, gab1, gag1, gabe1, gaW2, gab2, gaW3, gab3, GS);

  // ---- decoder / anomaly MLP heads (fused WMMA MLPs) ----
  // node_dec: 32 -> 128 -> 64 -> 256
  mlp_kernel<32, 128, 64, 256, false, false><<<469, 128, 0, stream>>>(
      Z, nullptr, nullptr, nullptr, Nn, HND1, ndb1, ndg1, ndbe1, HND2, ndb2, ndW3, HND3, ndb3, NR);
  // node_an: 32 -> 64 -> 32 -> 1
  mlp_kernel<32, 64, 32, 1, false, false><<<469, 128, 0, stream>>>(
      Z, nullptr, nullptr, nullptr, Nn, HNA1, nab1, nag1, nabe1, HNA2, nab2, naW3, nullptr, nab3, NS);
  // edge_dec: 96 -> 128 -> 64 -> 1, sigmoid (ef gathered on the fly)
  mlp_kernel<96, 128, 64, 1, true, true><<<7500, 128, 0, stream>>>(
      Z, srcA, dstA, eat, Ee, HED1, edb1, edg1, edbe1, HED2, edb2, edW3, nullptr, edb3, ER);
  // edge_an: 96 -> 64 -> 32 -> 1
  mlp_kernel<96, 64, 32, 1, false, true><<<7500, 128, 0, stream>>>(
      Z, srcA, dstA, eat, Ee, HEA1, eab1, eag1, eabe1, HEA2, eab2, eaW3, nullptr, eab3, ES);
}
